// MXMNet_32057635897563
// MI455X (gfx1250) — matrix operand order repaired and checked by reading, not compiled, not run on hardware
//
#include <hip/hip_runtime.h>
#include <hip/hip_bf16.h>
#include <cstddef>

typedef __attribute__((ext_vector_type(16))) __bf16 v16bf;
typedef __attribute__((ext_vector_type(8)))  __bf16 v8bf;
typedef __attribute__((ext_vector_type(4)))  __bf16 v4bf;
typedef __attribute__((ext_vector_type(8)))  float  v8f;
typedef __attribute__((ext_vector_type(4)))  float  v4f;

#define DIM      128
#define NNODES   5000
#define EL       20000
#define EG       60000
#define NTRIP    80000
#define NGRAPHS  64
#define NLAYER   6

#define LAYER_STRIDE 609793u      // f32 param elements per layer (pytree flatten)
#define WLAYER       606208u      // packed bf16 W elements per layer

#define TILE_M  32
#define A_PITCH 40                // bf16 elems; 80B row stride (16B aligned, bank-friendly)
#define B_PITCH 40                // bf16 elems per column slot in LDS

#define USE_ASYNC_LDS 1

// ---------------------------------------------------------------------------
// compile-time parameter layout (jax pytree flatten: dict keys sorted)
// ---------------------------------------------------------------------------
struct LO   { unsigned wq, b; };
struct Leaf { unsigned src, dst, kdim; };
#define NLEAF 31

struct LayerOffs {
  LO g_h, g_mlp, g_r1[2], g_r2[2], g_r3[2], g_xe;
  unsigned g_lin;
  LO l_h, l_ji1, l_ji2, l_jj, l_kj, l_sb1[2], l_sb2[2], l_r1[2], l_r2[2], l_r3[2], l_y[3];
  unsigned l_rbf1, l_rbf2, l_rbfo;
  unsigned yW, yWb;
  Leaf leaf[NLEAF];
  unsigned stotal, wtotal;
};
struct Walk { unsigned so, dof; int nl; };

constexpr unsigned addW(LayerOffs& r, Walk& w, unsigned kdim) {
  r.leaf[w.nl].src = w.so; r.leaf[w.nl].dst = w.dof; r.leaf[w.nl].kdim = kdim; w.nl++;
  unsigned d = w.dof; w.so += kdim * 128u; w.dof += kdim * 128u; return d;
}
constexpr LO addLin(LayerOffs& r, Walk& w, unsigned kdim) {
  LO t{}; t.wq = addW(r, w, kdim); t.b = w.so; w.so += 128u; return t;
}
constexpr LayerOffs make_offs() {
  LayerOffs r{}; Walk w{0u, 0u, 0};
  r.g_h = addLin(r, w, 128);
  r.g_lin = addW(r, w, 128);
  r.g_mlp = addLin(r, w, 128);
  for (int i = 0; i < 2; ++i) r.g_r1[i] = addLin(r, w, 128);
  for (int i = 0; i < 2; ++i) r.g_r2[i] = addLin(r, w, 128);
  for (int i = 0; i < 2; ++i) r.g_r3[i] = addLin(r, w, 128);
  r.g_xe = addLin(r, w, 384);
  r.l_h = addLin(r, w, 128);
  r.l_rbf1 = addW(r, w, 128); r.l_rbf2 = addW(r, w, 128); r.l_rbfo = addW(r, w, 128);
  r.l_ji1 = addLin(r, w, 384); r.l_ji2 = addLin(r, w, 128);
  r.l_jj  = addLin(r, w, 128); r.l_kj  = addLin(r, w, 384);
  for (int i = 0; i < 2; ++i) r.l_sb1[i] = addLin(r, w, 128);
  for (int i = 0; i < 2; ++i) r.l_sb2[i] = addLin(r, w, 128);
  for (int i = 0; i < 2; ++i) r.l_r1[i] = addLin(r, w, 128);
  for (int i = 0; i < 2; ++i) r.l_r2[i] = addLin(r, w, 128);
  for (int i = 0; i < 2; ++i) r.l_r3[i] = addLin(r, w, 128);
  r.yW = w.so; w.so += 128u; r.yWb = w.so; w.so += 1u;
  for (int i = 0; i < 3; ++i) r.l_y[i] = addLin(r, w, 128);
  r.stotal = w.so; r.wtotal = w.dof;
  return r;
}
constexpr LayerOffs OFFC = make_offs();
static_assert(OFFC.stotal == LAYER_STRIDE, "param walk mismatch");
static_assert(OFFC.wtotal == WLAYER, "packed walk mismatch");

// ---------------------------------------------------------------------------
// weight repack: f32 params -> bf16, chunk-major column-contiguous
// ---------------------------------------------------------------------------
__global__ void repack_w_kernel(const float* __restrict__ pf, __bf16* __restrict__ wq) {
  const unsigned L = blockIdx.y;
  for (unsigned d = blockIdx.x * blockDim.x + threadIdx.x; d < WLAYER;
       d += gridDim.x * blockDim.x) {
    unsigned src = 0, base = 0;
#pragma unroll
    for (int s = 0; s < NLEAF; ++s)
      if (d >= OFFC.leaf[s].dst) { src = OFFC.leaf[s].src; base = OFFC.leaf[s].dst; }
    const unsigned r = d - base;
    const unsigned kc = r >> 12, rem = r & 4095u, c = rem >> 5, kl = rem & 31u;
    const unsigned k = (kc << 5) + kl;
    wq[(size_t)L * WLAYER + d] =
        (__bf16)pf[(size_t)L * LAYER_STRIDE + src + (size_t)k * 128u + c];
  }
}

// ---------------------------------------------------------------------------
// vectorized utility kernels
// ---------------------------------------------------------------------------
__global__ void zero4_kernel(float4* __restrict__ p, long n4) {
  long i = (long)blockIdx.x * blockDim.x + threadIdx.x;
  long s = (long)gridDim.x * blockDim.x;
  const float4 z = {0.f, 0.f, 0.f, 0.f};
  for (; i < n4; i += s) p[i] = z;
}
__global__ void copy4_kernel(float4* __restrict__ d, const float4* __restrict__ s4, long n4) {
  long i = (long)blockIdx.x * blockDim.x + threadIdx.x;
  long s = (long)gridDim.x * blockDim.x;
  for (; i < n4; i += s) d[i] = s4[i];
}
// out[idx[e], :] += x[e, :]   (n4 = E*32)
__global__ void scatter_rows_kernel(const float* __restrict__ x, const int* __restrict__ idx,
                                    float* __restrict__ out, long n4) {
  long i = (long)blockIdx.x * blockDim.x + threadIdx.x;
  long s = (long)gridDim.x * blockDim.x;
  for (; i < n4; i += s) {
    long e = i >> 5; int c4 = (int)(i & 31) << 2;
    const float4 v = *(const float4*)(x + (e << 7) + c4);
    float* o = out + (size_t)idx[e] * DIM + c4;
    unsafeAtomicAdd(o + 0, v.x); unsafeAtomicAdd(o + 1, v.y);
    unsafeAtomicAdd(o + 2, v.z); unsafeAtomicAdd(o + 3, v.w);
  }
}
// out[sidx[t], :] += src[gidx[t], :] * sb[t, :]   (n4 = T*32)
__global__ void gather_mul_scatter_kernel(const float* __restrict__ src, const int* __restrict__ gidx,
                                          const float* __restrict__ sb, const int* __restrict__ sidx,
                                          float* __restrict__ out, long n4) {
  long i = (long)blockIdx.x * blockDim.x + threadIdx.x;
  long s = (long)gridDim.x * blockDim.x;
  for (; i < n4; i += s) {
    long t = i >> 5; int c4 = (int)(i & 31) << 2;
    const float4 a = *(const float4*)(src + (size_t)gidx[t] * DIM + c4);
    const float4 b = *(const float4*)(sb + (t << 7) + c4);
    float* o = out + (size_t)sidx[t] * DIM + c4;
    unsafeAtomicAdd(o + 0, a.x * b.x); unsafeAtomicAdd(o + 1, a.y * b.y);
    unsafeAtomicAdd(o + 2, a.z * b.z); unsafeAtomicAdd(o + 3, a.w * b.w);
  }
}
__global__ void gemv_acc_kernel(const float* __restrict__ X, const float* __restrict__ W,
                                const float* __restrict__ b, float* __restrict__ ns, int M) {
  int r = blockIdx.x * blockDim.x + threadIdx.x;
  if (r >= M) return;
  const v4f* x = (const v4f*)(X + (size_t)r * DIM);
  const v4f* w = (const v4f*)W;
  float a = b[0];
#pragma unroll
  for (int c = 0; c < DIM / 4; ++c) {
    v4f xv = x[c], wv = w[c];
    a += xv.x * wv.x + xv.y * wv.y + xv.z * wv.z + xv.w * wv.w;
  }
  ns[r] += a;
}
__global__ void batch_sum_kernel(const float* __restrict__ ns, const int* __restrict__ batch,
                                 float* __restrict__ out, int n) {
  int i = blockIdx.x * blockDim.x + threadIdx.x;
  if (i < n) unsafeAtomicAdd(&out[batch[i]], ns[i]);
}

// ---------------------------------------------------------------------------
// WMMA GEMM machinery
// ---------------------------------------------------------------------------
__device__ __forceinline__ float silu_f(float x) { return x / (1.0f + __expf(-x)); }

__device__ __forceinline__ v16bf lds_ld16(const __bf16* p) {
  v8bf lo = *(const v8bf*)p;
  v8bf hi = *(const v8bf*)(p + 8);
  return __builtin_shufflevector(lo, hi, 0,1,2,3,4,5,6,7,8,9,10,11,12,13,14,15);
}

// stage one 32x128 bf16 weight chunk (column-contiguous in global)
__device__ __forceinline__ void stage_B_chunk(const __bf16* __restrict__ Wc, __bf16* Bs) {
  const int t = threadIdx.x;
  const int c = t >> 1, hk = (t & 1) << 4;
#if USE_ASYNC_LDS
  unsigned lds = (unsigned)(uintptr_t)(Bs + c * B_PITCH + hk);
  unsigned long long ga = (unsigned long long)(uintptr_t)(Wc + c * 32 + hk);
  // async DMA: LDS[lds+off] = MEM[ga+off], tracked by ASYNCcnt (no VGPR round-trip)
  asm volatile("global_load_async_to_lds_b128 %0, %1, off"           :: "v"(lds), "v"(ga) : "memory");
  asm volatile("global_load_async_to_lds_b128 %0, %1, off offset:16" :: "v"(lds), "v"(ga) : "memory");
#else
  const uint4* s = (const uint4*)(Wc + c * 32 + hk);
  uint4* d = (uint4*)(Bs + c * B_PITCH + hk);
  d[0] = s[0]; d[1] = s[1];
#endif
}

__device__ __forceinline__ void async_fence() {
#if USE_ASYNC_LDS
  asm volatile("s_wait_asynccnt 0x0" ::: "memory");
#endif
}

// store 4 converted bf16 into A tile (k order: contiguous 0-31)
__device__ __forceinline__ void stage_A4(__bf16* As, int ar, int ac, v4f v) {
  const int j = ac;
  *(v4bf*)(As + ar * A_PITCH + j) = __builtin_convertvector(v, v4bf);
}

__device__ __forceinline__ void wmma_step(const __bf16* As, const __bf16* Bs,
                                          int m_half, int nbase, v8f& acc0, v8f& acc1) {
  const int lane  = threadIdx.x & 31;
  const int half16 = (lane >> 4) << 4;
  const int mr    = (lane & 15) + (m_half << 4);
  const int n     = lane & 15;
  v16bf a  = lds_ld16(As + mr * A_PITCH + half16);
  v16bf b0 = lds_ld16(Bs + (nbase + n) * B_PITCH + half16);
  v16bf b1 = lds_ld16(Bs + (nbase + 16 + n) * B_PITCH + half16);
  acc0 = __builtin_amdgcn_wmma_f32_16x16x32_bf16(false, a, false, b0, (short)0, acc0, false, false);
  acc1 = __builtin_amdgcn_wmma_f32_16x16x32_bf16(false, a, false, b1, (short)0, acc1, false, false);
}

__device__ __forceinline__ void gemm_epilogue(v8f acc0, v8f acc1, float* __restrict__ Y,
                                              const float* bias, const float* mulsrc,
                                              const float* addsrc, int act,
                                              int row0, int m_half, int nbase, int M) {
  const int lane = threadIdx.x & 31;
  const int half = lane >> 4;
  const int n    = lane & 15;
#pragma unroll
  for (int v = 0; v < 8; ++v) {
    const int row = row0 + m_half * 16 + v + 8 * half;
    if (row >= M) continue;
#pragma unroll
    for (int t = 0; t < 2; ++t) {
      const int col = nbase + t * 16 + n;
      float x = t ? acc1[v] : acc0[v];
      if (bias) x += bias[col];
      if (act)  x = silu_f(x);
      const size_t o = (size_t)row * DIM + col;
      if (mulsrc) x *= mulsrc[o];
      if (addsrc) x += addsrc[o];
      Y[o] = x;
    }
  }
}

__global__ __launch_bounds__(256)
void gemm128_kernel(const float* __restrict__ X, const __bf16* __restrict__ Wq,
                    const float* __restrict__ bias, const float* __restrict__ mulsrc,
                    const float* __restrict__ addsrc, float* __restrict__ Y, int M, int act) {
  __shared__ __bf16 As[2][TILE_M * A_PITCH];
  __shared__ __bf16 Bs[2][128 * B_PITCH];
  const int row0 = blockIdx.x * TILE_M;
  const int tid  = threadIdx.x;
  const int ar = tid >> 3, ac = (tid & 7) << 2;
  const int wid = tid >> 5;
  const int m_half = wid >> 2, nbase = (wid & 3) << 5;
  // OOB rows clamp to a valid row: their D rows are masked at the epilogue.
  const int arow = (row0 + ar < M) ? (row0 + ar) : (M - 1);
  const float* aptr = X + (size_t)arow * DIM + ac;
  v8f acc0 = {}, acc1 = {};
  // prologue: stage chunk 0 into buffer 0
  stage_B_chunk(Wq, Bs[0]);
  stage_A4(As[0], ar, ac, *(const v4f*)aptr);
  async_fence();
  __syncthreads();
#pragma unroll
  for (int kc = 0; kc < 4; ++kc) {
    const int cur = kc & 1, nxt = cur ^ 1;
    v4f a_next = {};
    if (kc + 1 < 4) {                       // issue next chunk's loads before compute
      a_next = *(const v4f*)(aptr + ((kc + 1) << 5));
      stage_B_chunk(Wq + ((kc + 1) << 12), Bs[nxt]);
    }
    wmma_step(As[cur], Bs[cur], m_half, nbase, acc0, acc1);
    if (kc + 1 < 4) {
      stage_A4(As[nxt], ar, ac, a_next);    // loadcnt wait lands after WMMAs issued
      async_fence();                        // B DMA had the WMMA phase to complete
      __syncthreads();                      // one barrier per chunk
    }
  }
  gemm_epilogue(acc0, acc1, Y, bias, mulsrc, addsrc, act, row0, m_half, nbase, M);
}

// K = 384; A row = concat( h[idxA[row]], h[idxB[row]], rbf[row] ), gathered in-tile
__global__ __launch_bounds__(256)
void gemm384_cat_kernel(const float* __restrict__ h, const int* __restrict__ idxA,
                        const int* __restrict__ idxB, const float* __restrict__ rbf,
                        const __bf16* __restrict__ Wq, const float* __restrict__ bias,
                        const float* __restrict__ mulsrc, const float* __restrict__ addsrc,
                        float* __restrict__ Y, int M) {
  __shared__ __bf16 As[2][TILE_M * A_PITCH];
  __shared__ __bf16 Bs[2][128 * B_PITCH];
  const int row0 = blockIdx.x * TILE_M;
  const int tid  = threadIdx.x;
  const int ar = tid >> 3, ac = (tid & 7) << 2;
  const int wid = tid >> 5;
  const int m_half = wid >> 2, nbase = (wid & 3) << 5;
  const int arow = (row0 + ar < M) ? (row0 + ar) : (M - 1);
  const int ia = idxA[arow], ib = idxB[arow];
  const float* s0 = h + (size_t)ia * DIM + ac;
  const float* s1 = h + (size_t)ib * DIM + ac;
  const float* s2 = rbf + (size_t)arow * DIM + ac;
  v8f acc0 = {}, acc1 = {};
  stage_B_chunk(Wq, Bs[0]);
  stage_A4(As[0], ar, ac, *(const v4f*)s0);
  async_fence();
  __syncthreads();
#pragma unroll
  for (int kc = 0; kc < 12; ++kc) {
    const int cur = kc & 1, nxt = cur ^ 1;
    v4f a_next = {};
    if (kc + 1 < 12) {
      const int kn  = kc + 1;
      const int sel = kn >> 2;
      const int kk  = (kn & 3) << 5;
      const float* sp = (sel == 0) ? s0 : ((sel == 1) ? s1 : s2);
      a_next = *(const v4f*)(sp + kk);
      stage_B_chunk(Wq + (kn << 12), Bs[nxt]);
    }
    wmma_step(As[cur], Bs[cur], m_half, nbase, acc0, acc1);
    if (kc + 1 < 12) {
      stage_A4(As[nxt], ar, ac, a_next);
      async_fence();
      __syncthreads();
    }
  }
  gemm_epilogue(acc0, acc1, Y, bias, mulsrc, addsrc, 1, row0, m_half, nbase, M);
}

// ---------------------------------------------------------------------------
extern "C" void kernel_launch(void* const* d_in, const int* in_sizes, int n_in,
                              void* d_out, int out_size, void* d_ws, size_t ws_size,
                              hipStream_t stream) {
  (void)in_sizes; (void)n_in; (void)out_size; (void)ws_size;
  const float* h_in   = (const float*)d_in[0];
  const float* rbfg   = (const float*)d_in[1];
  const float* rbfl   = (const float*)d_in[2];
  const float* sbf1in = (const float*)d_in[3];
  const float* sbf2in = (const float*)d_in[4];
  const float* pf     = (const float*)d_in[5];
  const int* eig      = (const int*)d_in[6];
  const int* eil      = (const int*)d_in[7];
  const int* idx_kj   = (const int*)d_in[8];
  const int* idx_ji_1 = (const int*)d_in[9];
  const int* idx_jj   = (const int*)d_in[10];
  const int* idx_ji_2 = (const int*)d_in[11];
  const int* batch    = (const int*)d_in[12];
  const int* rowg = eig;   const int* colg = eig + EG;
  const int* lj   = eil;   const int* li   = eil + EL;
  float* out = (float*)d_out;

  size_t woff = 0;
  auto carve = [&](size_t bytes) -> void* {
    void* p = (char*)d_ws + woff; woff += (bytes + 255) & ~(size_t)255; return p;
  };
  __bf16* wq = (__bf16*)carve((size_t)NLAYER * WLAYER * sizeof(__bf16));
  const size_t NB = (size_t)NNODES * DIM * 4;
  float* hA  = (float*)carve(NB);
  float* hB  = (float*)carve(NB);
  float* hC  = (float*)carve(NB);
  float* yb  = (float*)carve(NB);
  float* ns  = (float*)carve((size_t)NNODES * 4);
  float* egA = (float*)carve((size_t)EG * DIM * 4);
  float* egB = (float*)carve((size_t)EG * DIM * 4);
  float* elA = (float*)carve((size_t)EL * DIM * 4);
  float* elB = (float*)carve((size_t)EL * DIM * 4);
  float* elC = (float*)carve((size_t)EL * DIM * 4);
  float* sb  = (float*)carve((size_t)NTRIP * DIM * 4);

  int curL = 0;
  auto wqp = [&](unsigned rel) { return wq + (size_t)curL * WLAYER + rel; };
  auto bp  = [&](unsigned rel) { return pf + (size_t)curL * LAYER_STRIDE + rel; };
  auto g128L = [&](const float* X, const LO& o, const float* mul, const float* add,
                   float* Y, int M, int act) {
    gemm128_kernel<<<(M + TILE_M - 1) / TILE_M, 256, 0, stream>>>(
        X, wqp(o.wq), bp(o.b), mul, add, Y, M, act);
  };
  auto g128W = [&](const float* X, unsigned wrel, const float* mul, const float* add,
                   float* Y, int M) {
    gemm128_kernel<<<(M + TILE_M - 1) / TILE_M, 256, 0, stream>>>(
        X, wqp(wrel), nullptr, mul, add, Y, M, 0);
  };
  auto g384 = [&](const float* h, const int* ia, const int* ib, const float* rbf,
                  const LO& o, const float* mul, const float* add, float* Y, int M) {
    gemm384_cat_kernel<<<(M + TILE_M - 1) / TILE_M, 256, 0, stream>>>(
        h, ia, ib, rbf, wqp(o.wq), bp(o.b), mul, add, Y, M);
  };
  auto zero = [&](float* p, long n) { zero4_kernel<<<512, 256, 0, stream>>>((float4*)p, n >> 2); };
  auto copy = [&](float* d, const float* s, long n) {
    copy4_kernel<<<512, 256, 0, stream>>>((float4*)d, (const float4*)s, n >> 2);
  };

  repack_w_kernel<<<dim3(512, NLAYER), 256, 0, stream>>>(pf, wq);
  zero(ns, NNODES);
  copy(hA, h_in, (long)NNODES * DIM);

  for (curL = 0; curL < NLAYER; ++curL) {
    const LayerOffs& F = OFFC;

    // ---------------- global_mp: h in hA (= res_h) ----------------
    g128L(hA, F.g_h, nullptr, nullptr, hB, NNODES, 1);                    // h1
    g384(hB, rowg, colg, rbfg, F.g_xe, nullptr, nullptr, egA, EG);        // x_edge
    g128W(rbfg, F.g_lin, egA, nullptr, egB, EG);                          // msg
    copy(hC, hB, (long)NNODES * DIM);
    scatter_rows_kernel<<<((long)EG * 32 + 255) / 256, 256, 0, stream>>>(egB, rowg, hC, (long)EG * 32);
    g128L(hC, F.g_r1[0], nullptr, nullptr, hB, NNODES, 1);                // res1
    g128L(hB, F.g_r1[1], nullptr, hC, hC, NNODES, 1);
    g128L(hC, F.g_mlp, nullptr, hA, hC, NNODES, 1);                       // mlp + res_h
    g128L(hC, F.g_r2[0], nullptr, nullptr, hB, NNODES, 1);                // res2
    g128L(hB, F.g_r2[1], nullptr, hC, hC, NNODES, 1);
    g128L(hC, F.g_r3[0], nullptr, nullptr, hB, NNODES, 1);                // res3
    g128L(hB, F.g_r3[1], nullptr, hC, hC, NNODES, 1);
    g384(hC, rowg, colg, rbfg, F.g_xe, nullptr, nullptr, egA, EG);        // propagate 2
    g128W(rbfg, F.g_lin, egA, nullptr, egB, EG);
    copy(hB, hC, (long)NNODES * DIM);
    scatter_rows_kernel<<<((long)EG * 32 + 255) / 256, 256, 0, stream>>>(egB, rowg, hB, (long)EG * 32);

    // ---------------- local_mp (res_h in hB) ----------------
    g128L(hB, F.l_h, nullptr, nullptr, hA, NNODES, 1);                    // h1
    g128W(rbfl, F.l_rbf1, nullptr, nullptr, elB, EL);                     // rl1
    g384(hA, li, lj, rbfl, F.l_kj, elB, nullptr, elA, EL);                // m_kj
    g128L(sbf1in, F.l_sb1[0], nullptr, nullptr, sb, NTRIP, 1);            // sbf1 mlp
    g128L(sb,     F.l_sb1[1], nullptr, nullptr, sb, NTRIP, 1);
    zero(elC, (long)EL * DIM);
    gather_mul_scatter_kernel<<<((long)NTRIP * 32 + 255) / 256, 256, 0, stream>>>(
        elA, idx_kj, sb, idx_ji_1, elC, (long)NTRIP * 32);
    g384(hA, li, lj, rbfl, F.l_ji1, nullptr, elC, elB, EL);               // m1
    g128W(rbfl, F.l_rbf2, nullptr, nullptr, elA, EL);                     // rl2
    g128L(elB, F.l_jj, elA, nullptr, elA, EL, 1);                         // m_jj
    g128L(sbf2in, F.l_sb2[0], nullptr, nullptr, sb, NTRIP, 1);            // sbf2 mlp
    g128L(sb,     F.l_sb2[1], nullptr, nullptr, sb, NTRIP, 1);
    zero(elC, (long)EL * DIM);
    gather_mul_scatter_kernel<<<((long)NTRIP * 32 + 255) / 256, 256, 0, stream>>>(
        elA, idx_jj, sb, idx_ji_2, elC, (long)NTRIP * 32);
    g128L(elB, F.l_ji2, nullptr, elC, elB, EL, 1);                        // m2
    g128W(rbfl, F.l_rbfo, elB, nullptr, elA, EL);                         // m
    zero(hC, (long)NNODES * DIM);
    scatter_rows_kernel<<<((long)EL * 32 + 255) / 256, 256, 0, stream>>>(elA, li, hC, (long)EL * 32);
    g128L(hC, F.l_r1[0], nullptr, nullptr, hA, NNODES, 1);                // res1
    g128L(hA, F.l_r1[1], nullptr, hC, hC, NNODES, 1);
    g128L(hC, F.l_h, nullptr, hB, hC, NNODES, 1);                         // h_mlp + res_h
    g128L(hC, F.l_r2[0], nullptr, nullptr, hA, NNODES, 1);                // res2
    g128L(hA, F.l_r2[1], nullptr, hC, hC, NNODES, 1);
    g128L(hC, F.l_r3[0], nullptr, nullptr, hA, NNODES, 1);                // res3
    g128L(hA, F.l_r3[1], nullptr, hC, hC, NNODES, 1);
    g128L(hC, F.l_y[0], nullptr, nullptr, yb, NNODES, 1);                 // y head
    g128L(yb, F.l_y[1], nullptr, nullptr, yb, NNODES, 1);
    g128L(yb, F.l_y[2], nullptr, nullptr, yb, NNODES, 1);
    gemv_acc_kernel<<<(NNODES + 255) / 256, 256, 0, stream>>>(
        yb, pf + (size_t)curL * LAYER_STRIDE + F.yW, pf + (size_t)curL * LAYER_STRIDE + F.yWb,
        ns, NNODES);
    copy(hA, hC, (long)NNODES * DIM);   // next layer expects h in hA
  }

  zero(out, NGRAPHS);
  batch_sum_kernel<<<(NNODES + 255) / 256, 256, 0, stream>>>(ns, batch, out, NNODES);
}
